// LambdaLayer3d_10411000725998
// MI455X (gfx1250) — matrix operand order repaired and checked by reading, not compiled, not run on hardware
//
#include <hip/hip_runtime.h>
#include <hip/hip_bf16.h>

// LambdaLayer3d for MI455X (gfx1250): fused positional-lambda via bf16 WMMA.
// Sizes fixed by the reference: B=4, C=64, K=16, H=4, V=16, U=1, M=14, n=2744.

#define BDIM   4
#define CDIM   64
#define NKK    16
#define NHEADS 4
#define NVV    16
#define NPOS   2744          // 14^3
#define NPAD   2752          // 16-aligned (172 tiles of 16, 86 tiles of 32)
#define RELC   9841          // 13*(729+27+1): shift so off = P[m]-P[i]+RELC in [0,27^3)
#define RPESZ  (27 * 27 * 27 * 16)

typedef __attribute__((ext_vector_type(16))) __bf16 v16bf;
typedef __attribute__((ext_vector_type(8)))  float  v8f;

__device__ __forceinline__ unsigned short f2bf_rne(float f) {
  unsigned int u = __float_as_uint(f);
  u += 0x7FFFu + ((u >> 16) & 1u);          // round-to-nearest-even
  return (unsigned short)(u >> 16);
}

// ---------------------------------------------------------------- init: P table + bf16 v pad
__global__ void init_kernel(int* __restrict__ P, unsigned short* __restrict__ vbf) {
  int m = blockIdx.x * blockDim.x + threadIdx.x;
  if (m >= NPAD) return;
  int mm = m < NPOS ? m : (NPOS - 1);
  int w = mm % 14; int t = mm / 14; int h = t % 14; int d = t / 14;
  P[m] = (d * 27 + h) * 27 + w;             // base-27 linearized position
  if (m >= NPOS) {                          // zero K-padding rows of bf16 V
    for (int r = 0; r < BDIM * NVV; ++r) vbf[(size_t)r * NPAD + m] = 0;
  }
}

// ---------------------------------------------------------------- one-time rel_pos_emb -> bf16
__global__ __launch_bounds__(256) void cvt_rpe_kernel(const float* __restrict__ rpe,
                                                      unsigned short* __restrict__ rpe16) {
  int idx = blockIdx.x * 256 + threadIdx.x;
  if (idx < RPESZ) rpe16[idx] = f2bf_rne(rpe[idx]);
}

// ---------------------------------------------------------------- q/k/v 1x1x1 conv projections
__global__ __launch_bounds__(256) void proj_kernel(
    const float* __restrict__ x,
    const float* __restrict__ Wq, const float* __restrict__ Wk, const float* __restrict__ Wv,
    float* __restrict__ qb, float* __restrict__ kb, float* __restrict__ vb) {
  __shared__ float sW[96 * 64];             // 24 KB of the 320 KB WGP LDS
  for (int idx = threadIdx.x; idx < 96 * 64; idx += 256) {
    float w;
    if (idx < 4096)      w = Wq[idx];
    else if (idx < 5120) w = Wk[idx - 4096];
    else                 w = Wv[idx - 5120];
    sW[idx] = w;
  }
  __syncthreads();
  int t = blockIdx.x * 256 + threadIdx.x;
  if (t >= BDIM * NPOS) return;
  int b = t / NPOS, pos = t - b * NPOS;
  float xi[64];
  const float* xp = x + (size_t)b * 64 * NPOS + pos;
#pragma unroll
  for (int c = 0; c < 64; ++c) xi[c] = xp[(size_t)c * NPOS];
  for (int o = 0; o < 64; ++o) {
    const float* w = &sW[o * 64]; float s = 0.f;
#pragma unroll
    for (int c = 0; c < 64; ++c) s += w[c] * xi[c];
    qb[((size_t)b * 64 + o) * NPOS + pos] = s;
  }
  for (int o = 0; o < 16; ++o) {
    const float* w = &sW[4096 + o * 64]; float s = 0.f;
#pragma unroll
    for (int c = 0; c < 64; ++c) s += w[c] * xi[c];
    kb[((size_t)b * 16 + o) * NPOS + pos] = s;
  }
  for (int o = 0; o < 16; ++o) {
    const float* w = &sW[5120 + o * 64]; float s = 0.f;
#pragma unroll
    for (int c = 0; c < 64; ++c) s += w[c] * xi[c];
    vb[((size_t)b * 16 + o) * NPOS + pos] = s;
  }
}

// ---------------------------------------------------------------- BN stats: 64 q-ch + 16 v-ch
__global__ __launch_bounds__(256) void bn_stats_kernel(
    const float* __restrict__ qb, const float* __restrict__ vb,
    float* __restrict__ meanA, float* __restrict__ rstdA) {
  const int c = blockIdx.x;                 // 0..79
  const float* base; size_t bstride;
  if (c < 64) { base = qb + (size_t)c * NPOS;        bstride = (size_t)64 * NPOS; }
  else        { base = vb + (size_t)(c - 64) * NPOS; bstride = (size_t)16 * NPOS; }
  float s = 0.f, s2 = 0.f;
  for (int j = threadIdx.x; j < BDIM * NPOS; j += 256) {
    int b = j / NPOS, pos = j - b * NPOS;
    float xv = base[(size_t)b * bstride + pos];
    s += xv; s2 += xv * xv;
  }
  __shared__ float sh1[256], sh2[256];
  sh1[threadIdx.x] = s; sh2[threadIdx.x] = s2; __syncthreads();
  for (int off = 128; off > 0; off >>= 1) {
    if (threadIdx.x < off) {
      sh1[threadIdx.x] += sh1[threadIdx.x + off];
      sh2[threadIdx.x] += sh2[threadIdx.x + off];
    }
    __syncthreads();
  }
  if (threadIdx.x == 0) {
    float m  = sh1[0] / (float)(BDIM * NPOS);
    float vr = sh2[0] / (float)(BDIM * NPOS) - m * m;
    meanA[c] = m; rstdA[c] = rsqrtf(vr + 1e-5f);
  }
}

// ---------------------------------------------------------------- in-place softmax over m, per (b,kdim)
__global__ __launch_bounds__(256) void softmax_kernel(float* __restrict__ kb) {
  float* row = kb + (size_t)blockIdx.x * NPOS;
  __shared__ float sh[256];
  float mx = -3.4e38f;
  for (int j = threadIdx.x; j < NPOS; j += 256) mx = fmaxf(mx, row[j]);
  sh[threadIdx.x] = mx; __syncthreads();
  for (int off = 128; off > 0; off >>= 1) {
    if (threadIdx.x < off) sh[threadIdx.x] = fmaxf(sh[threadIdx.x], sh[threadIdx.x + off]);
    __syncthreads();
  }
  mx = sh[0]; __syncthreads();
  float s = 0.f;
  for (int j = threadIdx.x; j < NPOS; j += 256) s += expf(row[j] - mx);
  sh[threadIdx.x] = s; __syncthreads();
  for (int off = 128; off > 0; off >>= 1) {
    if (threadIdx.x < off) sh[threadIdx.x] += sh[threadIdx.x + off];
    __syncthreads();
  }
  float inv = 1.f / sh[0];
  for (int j = threadIdx.x; j < NPOS; j += 256) row[j] = expf(row[j] - mx) * inv;
}

// ---------------------------------------------------------------- BN apply (q,v in place) + bf16 V copy
__global__ __launch_bounds__(256) void bn_apply_kernel(
    float* __restrict__ qb, float* __restrict__ vb, unsigned short* __restrict__ vbf,
    const float* __restrict__ meanA, const float* __restrict__ rstdA,
    const float* __restrict__ gq, const float* __restrict__ bq,
    const float* __restrict__ gv, const float* __restrict__ bv) {
  int idx = blockIdx.x * 256 + threadIdx.x;
  if (idx >= BDIM * 80 * NPOS) return;
  int b   = idx / (80 * NPOS);
  int rem = idx - b * 80 * NPOS;
  int ch  = rem / NPOS;
  int pos = rem - ch * NPOS;
  if (ch < 64) {
    size_t o = ((size_t)b * 64 + ch) * NPOS + pos;
    float xv = qb[o];
    qb[o] = (xv - meanA[ch]) * rstdA[ch] * gq[ch] + bq[ch];
  } else {
    int cv = ch - 64;
    size_t o = ((size_t)b * 16 + cv) * NPOS + pos;
    float xv = vb[o];
    xv = (xv - meanA[ch]) * rstdA[ch] * gv[cv] + bv[cv];
    vb[o] = xv;
    vbf[((size_t)b * 16 + cv) * NPAD + pos] = f2bf_rne(xv);
  }
}

// ---------------------------------------------------------------- lambda_c[b,k,v] = sum_m ksm * vn
__global__ __launch_bounds__(256) void lambdac_kernel(
    const float* __restrict__ kb, const float* __restrict__ vb, float* __restrict__ lc) {
  int b = blockIdx.x;
  int kd = threadIdx.x >> 4, vv = threadIdx.x & 15;
  const float* kr = kb + ((size_t)b * 16 + kd) * NPOS;
  const float* vr = vb + ((size_t)b * 16 + vv) * NPOS;
  float s = 0.f;
  for (int m = 0; m < NPOS; ++m) s += kr[m] * vr[m];
  lc[((size_t)b * 16 + kd) * 16 + vv] = s;
}

// ---------------------------------------------------------------- main: fused lambda_p + y via WMMA
// One wave per query i. A(16k x 32m) gathered from bf16 rel_pos_emb with
// unsigned 32-bit offsets (SADDR+voffset addressing, no 64-bit VALU adds);
// B(32m x 16v) = aligned 32B loads from bf16 V. K-loop unrolled x2 so the
// scheduler can overlap next-tile gathers/loads with current-tile WMMAs.
// Accumulators start at lambda_c so y_c is fused; epilogue contracts with q.
__global__ __launch_bounds__(128) void lambda_pos_kernel(
    const unsigned short* __restrict__ rpe16, const float* __restrict__ qn,
    const unsigned short* __restrict__ vbf, const int* __restrict__ P,
    const float* __restrict__ lc, float* __restrict__ out) {
  const int lane          = threadIdx.x & 31;
  const int wave          = threadIdx.x >> 5;
  const int i             = blockIdx.x * 4 + wave;   // grid sized so i < NPOS exactly
  const unsigned int hi   = (unsigned)(lane >> 4);   // half-wave select
  const unsigned int lidx = (unsigned)(lane & 15);   // A row (k) and C/D column (v)
  const unsigned int c0   = (unsigned)(RELC - P[i]); // P[m]+c0 in [0, 27^3)

  v8f acc[4];
#pragma unroll
  for (int b = 0; b < 4; ++b)
#pragma unroll
    for (int r = 0; r < 8; ++r)
      acc[b][r] = lc[((size_t)b * 16 + r + hi * 8) * 16 + lidx];   // C init = lambda_c

#pragma unroll 2
  for (unsigned int mt = 0; mt < NPAD; mt += 32) {
    // A fragment (16x32 bf16): VGPR j holds K pair {2j,2j+1}
    // (+16 for lanes 16-31); rows striped over lanes (M = lane&15).
    union { unsigned int u[8]; v16bf v; } a;
#pragma unroll
    for (int j = 0; j < 8; ++j) {
      unsigned int Klo = 2u * j + hi * 16u;
      unsigned int m0  = mt + Klo;                 // even -> P pair is one b64 load
      int2 pp = *(const int2*)(P + m0);
      unsigned int lo = rpe16[((unsigned)pp.x + c0) * 16u + lidx];  // 32-bit voffset
      unsigned int hh = rpe16[((unsigned)pp.y + c0) * 16u + lidx];
      a.u[j] = lo | (hh << 16);
    }
    const unsigned int moff = mt + hi * 16u;  // B 32x16: lanes 0-15 K=0..15, 16-31 K=16..31
    v16bf bfr[4];
#pragma unroll
    for (int b = 0; b < 4; ++b)
      bfr[b] = *(const v16bf*)(vbf + (unsigned)(b * 16 + lidx) * NPAD + moff);
#pragma unroll
    for (int b = 0; b < 4; ++b)
      acc[b] = __builtin_amdgcn_wmma_f32_16x16x32_bf16(
          false, a.v, false, bfr[b], (short)0, acc[b], false, false);
  }

  // Epilogue: y[b, h*16+v, i] = sum_k qn[b, h*16+k, i] * acc[b][k,v]
#pragma unroll
  for (int b = 0; b < 4; ++b) {
#pragma unroll
    for (int h = 0; h < NHEADS; ++h) {
      float p = 0.f;
#pragma unroll
      for (int r = 0; r < 8; ++r) {
        int ch = h * 16 + r + hi * 8;       // this half-wave owns rows k = hi*8 .. hi*8+7
        p += qn[((size_t)b * 64 + ch) * NPOS + i] * acc[b][r];
      }
      p += __shfl_xor(p, 16, 32);           // combine the two 8-row halves
      if (lane < 16)
        out[((size_t)b * 64 + h * 16 + lidx) * NPOS + i] = p;
    }
  }
}

// ---------------------------------------------------------------- launcher
extern "C" void kernel_launch(void* const* d_in, const int* in_sizes, int n_in,
                              void* d_out, int out_size, void* d_ws, size_t ws_size,
                              hipStream_t stream) {
  const float* x   = (const float*)d_in[0];
  const float* Wq  = (const float*)d_in[1];
  const float* Wk  = (const float*)d_in[2];
  const float* Wv  = (const float*)d_in[3];
  const float* rpe = (const float*)d_in[4];
  const float* gq  = (const float*)d_in[5];
  const float* bq  = (const float*)d_in[6];
  const float* gv  = (const float*)d_in[7];
  const float* bv  = (const float*)d_in[8];
  float* out = (float*)d_out;

  char* p = (char*)d_ws;
  auto carve = [&](size_t bytes) -> char* {
    char* r = p; p += (bytes + 255) & ~(size_t)255; return r;
  };
  float*          qb    = (float*)carve(sizeof(float) * BDIM * CDIM * NPOS); // q, normalized in place
  float*          kb    = (float*)carve(sizeof(float) * BDIM * NKK  * NPOS); // k, softmaxed in place
  float*          vb    = (float*)carve(sizeof(float) * BDIM * NVV  * NPOS); // v, normalized in place
  unsigned short* vbf   = (unsigned short*)carve(sizeof(unsigned short) * BDIM * NVV * NPAD);
  unsigned short* rpe16 = (unsigned short*)carve(sizeof(unsigned short) * RPESZ);
  int*            P     = (int*)carve(sizeof(int) * NPAD);
  float*          mnA   = (float*)carve(sizeof(float) * 80);
  float*          rsA   = (float*)carve(sizeof(float) * 80);
  float*          lc    = (float*)carve(sizeof(float) * BDIM * NKK * NVV);

  init_kernel   <<<(NPAD + 255) / 256, 256, 0, stream>>>(P, vbf);
  cvt_rpe_kernel<<<(RPESZ + 255) / 256, 256, 0, stream>>>(rpe, rpe16);
  proj_kernel   <<<(BDIM * NPOS + 255) / 256, 256, 0, stream>>>(x, Wq, Wk, Wv, qb, kb, vb);
  bn_stats_kernel<<<80, 256, 0, stream>>>(qb, vb, mnA, rsA);
  softmax_kernel<<<BDIM * NKK, 256, 0, stream>>>(kb);
  bn_apply_kernel<<<(BDIM * 80 * NPOS + 255) / 256, 256, 0, stream>>>(qb, vb, vbf, mnA, rsA, gq, bq, gv, bv);
  lambdac_kernel<<<BDIM, 256, 0, stream>>>(kb, vb, lc);
  lambda_pos_kernel<<<NPOS / 4, 128, 0, stream>>>(rpe16, qb, vbf, P, lc, out);
}
